// KeypointDetector_28991029248088
// MI455X (gfx1250) — compile-verified, matrix operand order checked by reading, not run on hardware
//
#include <hip/hip_runtime.h>
#include <hip/hip_bf16.h>

typedef __attribute__((ext_vector_type(16))) _Float16 v16h;
typedef __attribute__((ext_vector_type(8)))  float    v8f;

#define GH 90
#define GW 160
#define HW (GH*GW)
#define PH (GH+2)   // zero-halo padded height (92)
#define PW (GW+2)   // zero-halo padded width  (162)
#define IC 256
#define W8 (GW*8)   // 1280
#define H8 (GH*8)   // 720
#define CAP 4096    // candidate cap for top-k

static __device__ __forceinline__ int iclamp(int v, int lo, int hi) {
    return v < lo ? lo : (v > hi ? hi : v);
}

// ---------------------------------------------------------------- zero fill
__global__ void kd_zero_f32(float* __restrict__ p, size_t n) {
    size_t i = (size_t)blockIdx.x * blockDim.x + threadIdx.x;
    if (i < n) p[i] = 0.0f;
}

// ------------------------------------------------------- weight repack (f16)
// conv1_w [256,256,3,3] f32 -> w1h [9][256][256] f16  (tap-major, O rows, I cols)
__global__ void kd_prep_w1(const float* __restrict__ w, _Float16* __restrict__ wh) {
    int i = blockIdx.x * blockDim.x + threadIdx.x;
    if (i >= 9 * 256 * 256) return;
    int tap = i / (256 * 256);
    int o   = (i / 256) % 256;
    int ci  = i % 256;
    wh[i] = (_Float16)w[(size_t)(o * 256 + ci) * 9 + tap];
}
// conv2_w [65,256,3,3] f32 -> w2h [9][80][256] f16, O padded 65->80 with zeros
__global__ void kd_prep_w2(const float* __restrict__ w, const float* __restrict__ b,
                           _Float16* __restrict__ wh, float* __restrict__ bpad) {
    int i = blockIdx.x * blockDim.x + threadIdx.x;
    if (i < 80) bpad[i] = (i < 65) ? b[i] : 0.0f;
    if (i >= 9 * 80 * 256) return;
    int tap = i / (80 * 256);
    int o   = (i / 256) % 80;
    int ci  = i % 256;
    wh[i] = (o < 65) ? (_Float16)w[(size_t)(o * 256 + ci) * 9 + tap] : (_Float16)0.0f;
}

// ------------------------------------------------------- event scatter (atomics)
// one wave per event: 8 float atomics per lane into fsum [B][HW][256], count [B][HW]
__global__ void kd_scatter(const float* __restrict__ feat, const float* __restrict__ pos,
                           const float* __restrict__ mask,
                           float* __restrict__ fsum, float* __restrict__ cnt, int B, int N) {
    int wid = blockIdx.x * (blockDim.x >> 5) + (threadIdx.x >> 5);
    if (wid >= B * N) return;
    int lane = threadIdx.x & 31;
    int b = wid / N, e = wid % N;
    size_t ev = (size_t)b * N + e;
    float px = pos[ev * 2 + 0];
    float py = pos[ev * 2 + 1];
    float m  = mask[ev];
    int xb = iclamp((int)(px * (float)(GW - 1)), 0, GW - 1);
    int yb = iclamp((int)(py * (float)(GH - 1)), 0, GH - 1);
    size_t cell = (size_t)b * HW + (size_t)yb * GW + xb;
    const float* f = feat + ev * IC;
    float* dst = fsum + cell * IC;
    #pragma unroll
    for (int c = lane; c < IC; c += 32)
        atomicAdd(dst + c, f[c] * m);
    if (lane == 0) atomicAdd(cnt + cell, m);
}

// -------------------------- normalize -> grid out (NCHW f32) + Xh (padded NHWC f16)
__global__ void kd_normalize(const float* __restrict__ fsum, const float* __restrict__ cnt,
                             float* __restrict__ gridOut, _Float16* __restrict__ Xh, int B) {
    size_t i = (size_t)blockIdx.x * blockDim.x + threadIdx.x;
    size_t total = (size_t)B * HW * IC;
    if (i >= total) return;
    int    c    = (int)(i & (IC - 1));
    size_t cell = i >> 8;             // b*HW + pixel
    int    b    = (int)(cell / HW);
    int    p    = (int)(cell % HW);
    int    y    = p / GW, x = p % GW;
    float cv = cnt[cell];
    cv = cv < 1.0f ? 1.0f : cv;
    float val = fsum[i] / cv;
    Xh[(((size_t)b * PH + (y + 1)) * PW + (x + 1)) * IC + c] = (_Float16)val;
    gridOut[((size_t)b * IC + c) * HW + p] = val;
}

// ------------------------------------------------------- implicit-GEMM 3x3 conv via WMMA
// X : [B,PH,PW,IC] f16 zero-haloed NHWC.  Wt : [9][OCPAD][IC] f16.  bias : [OCPAD] f32.
// One wave = 16(pixel) x 16(out-ch) tile; K = 9*IC in chunks of 32 -> 72 WMMAs.
// A 16x32 f16 frag (ISA 7.12.2): lane m = lane&15, group g = lane>>4 selects K half;
// halves 0..7 = k cb+8g+0..7, halves 8..15 = k cb+16+8g+0..7 -> exactly two b128 loads.
// B 32x16 frag mirrors A with n = lane&15 (weights stored one row per output channel).
template <int OCPAD, bool RELU, bool OUT_F16, bool OUT_PADDED>
__global__ __launch_bounds__(256)
void kd_conv3x3_wmma(const _Float16* __restrict__ X, const _Float16* __restrict__ Wt,
                     const float* __restrict__ bias, void* __restrict__ out, int B) {
    const int NT = OCPAD / 16;
    const int WT = GW / 16;     // 10 tiles per row
    const int MT = GH * WT;     // 900
    int wid = blockIdx.x * (blockDim.x >> 5) + (threadIdx.x >> 5);
    if (wid >= B * MT * NT) return;

    int lane  = threadIdx.x & 31;
    int g     = lane >> 4;
    int hlane = lane & 15;
    int b  = wid / (MT * NT);
    int r  = wid % (MT * NT);
    int mt = r / NT, nt = r % NT;
    int y  = mt / WT, x0 = (mt % WT) * 16;

    union Frag { v16h v; uint4 q[2]; };
    v8f acc = {0.f, 0.f, 0.f, 0.f, 0.f, 0.f, 0.f, 0.f};

    #pragma unroll
    for (int tap = 0; tap < 9; ++tap) {
        // padded coords: src (y+ky-1)+1 = y + tap/3 ; (x+kx-1)+1 = x + tap%3
        int yy = y + tap / 3;
        int xx = x0 + hlane + tap % 3;
        const uint4* ap = (const uint4*)(X + (((size_t)b * PH + yy) * PW + xx) * IC);
        const uint4* bp = (const uint4*)(Wt + ((size_t)tap * OCPAD + nt * 16 + hlane) * IC);
        if (tap < 8)  // warm next tap's weight row into cache
            __builtin_prefetch((const char*)bp + (size_t)OCPAD * IC * 2, 0, 2);
        #pragma unroll
        for (int cb = 0; cb < IC; cb += 32) {
            Frag a, bf;
            a.q[0]  = ap[(cb >> 3) + g];
            a.q[1]  = ap[(cb >> 3) + g + 2];
            bf.q[0] = bp[(cb >> 3) + g];
            bf.q[1] = bp[(cb >> 3) + g + 2];
            acc = __builtin_amdgcn_wmma_f32_16x16x32_f16(
                false, a.v, false, bf.v, (short)0, acc, false, false);
        }
    }

    // D layout: lane n = lane&15, row m = 8*g + v
    int n = nt * 16 + hlane;
    float bs = bias[n];
    #pragma unroll
    for (int v = 0; v < 8; ++v) {
        int m = 8 * g + v;
        int x = x0 + m;
        float rv = acc[v] + bs;
        if (RELU) rv = rv > 0.0f ? rv : 0.0f;
        size_t o;
        if constexpr (OUT_PADDED)
            o = (((size_t)b * PH + (y + 1)) * PW + (x + 1)) * OCPAD + n;
        else
            o = (((size_t)b * GH + y) * GW + x) * OCPAD + n;
        if constexpr (OUT_F16) ((_Float16*)out)[o] = (_Float16)rv;
        else                   ((float*)out)[o]    = rv;
    }
}

// ------------------------------------------------------- grouped softmax -> heat scores
// Z : [B,GH,GW,80] f32.  softmax over i of Z[.., i*8+j]; heat[b][(y*8+i)*1280 + x*8+j]
__global__ void kd_heat(const float* __restrict__ Z, float* __restrict__ scores, int B) {
    int i = blockIdx.x * blockDim.x + threadIdx.x;
    if (i >= B * HW) return;
    int b = i / HW, p = i % HW;
    int y = p / GW, x = p % GW;
    const float* z = Z + (size_t)i * 80;
    float* sb = scores + (size_t)b * (H8 * W8);
    for (int j = 0; j < 8; ++j) {
        float mx = -1e30f;
        #pragma unroll
        for (int ii = 0; ii < 8; ++ii) mx = fmaxf(mx, z[ii * 8 + j]);
        float e[8], s = 0.0f;
        #pragma unroll
        for (int ii = 0; ii < 8; ++ii) { e[ii] = __expf(z[ii * 8 + j] - mx); s += e[ii]; }
        float inv = 1.0f / s;
        #pragma unroll
        for (int ii = 0; ii < 8; ++ii)
            sb[(size_t)(y * 8 + ii) * W8 + (x * 8 + j)] = e[ii] * inv;
    }
}

// ------------------------------------------------------- top-k via histogram threshold
__global__ void kd_hist(const float* __restrict__ scores, int* __restrict__ hist, int B) {
    size_t i = (size_t)blockIdx.x * blockDim.x + threadIdx.x;
    size_t total = (size_t)B * H8 * W8;
    if (i >= total) return;
    int b = (int)(i / (H8 * W8));
    int bin = iclamp((int)(scores[i] * 1024.0f), 0, 1023);
    atomicAdd(&hist[b * 1024 + bin], 1);
}

__global__ void kd_thresh(const int* __restrict__ hist, float* __restrict__ thresh,
                          int B, int topk) {
    int b = threadIdx.x;
    if (b >= B) return;
    int cum = 0, t = 0;
    for (int i = 1023; i >= 0; --i) {
        cum += hist[b * 1024 + i];
        if (cum >= topk) { t = i; break; }
    }
    thresh[b] = (float)t * (1.0f / 1024.0f);
}

__global__ void kd_compact(const float* __restrict__ scores, const float* __restrict__ thresh,
                           int* __restrict__ ccnt, float* __restrict__ cs,
                           int* __restrict__ ci, int B) {
    size_t i = (size_t)blockIdx.x * blockDim.x + threadIdx.x;
    size_t total = (size_t)B * H8 * W8;
    if (i >= total) return;
    int b = (int)(i / (H8 * W8));
    int p = (int)(i % (H8 * W8));
    float s = scores[i];
    if (s >= thresh[b]) {
        int pos = atomicAdd(&ccnt[b], 1);
        if (pos < CAP) { cs[(size_t)b * CAP + pos] = s; ci[(size_t)b * CAP + pos] = p; }
    }
}

__global__ void kd_topk(const int* __restrict__ ccnt, const float* __restrict__ cs,
                        const int* __restrict__ ci, float* __restrict__ kp_out,
                        float* __restrict__ sc_out, int topk) {
    __shared__ float s[CAP];
    __shared__ int   id[CAP];
    int b = blockIdx.x;
    int n = ccnt[b]; if (n > CAP) n = CAP;
    for (int i = threadIdx.x; i < CAP; i += blockDim.x) {
        s[i]  = (i < n) ? cs[(size_t)b * CAP + i] : -1e30f;
        id[i] = (i < n) ? ci[(size_t)b * CAP + i] : 0;
    }
    __syncthreads();
    for (int k = 2; k <= CAP; k <<= 1) {
        for (int j = k >> 1; j > 0; j >>= 1) {
            for (int t = threadIdx.x; t < CAP; t += blockDim.x) {
                int ixj = t ^ j;
                if (ixj > t) {
                    bool desc = ((t & k) == 0);
                    float a0 = s[t], a1 = s[ixj];
                    bool sw = desc ? (a0 < a1) : (a0 > a1);
                    if (sw) {
                        s[t] = a1; s[ixj] = a0;
                        int tmp = id[t]; id[t] = id[ixj]; id[ixj] = tmp;
                    }
                }
            }
            __syncthreads();
        }
    }
    for (int i = threadIdx.x; i < topk; i += blockDim.x) {
        int idx = id[i];
        float kx = (float)(idx % W8) * (1.0f / (float)W8);
        float ky = (float)(idx / W8) * (1.0f / (float)H8);
        kp_out[((size_t)b * topk + i) * 2 + 0] = kx;
        kp_out[((size_t)b * topk + i) * 2 + 1] = ky;
        sc_out[(size_t)b * topk + i] = s[i];
    }
}

// ================================================================ launcher
extern "C" void kernel_launch(void* const* d_in, const int* in_sizes, int n_in,
                              void* d_out, int out_size, void* d_ws, size_t ws_size,
                              hipStream_t stream) {
    const float* feat = (const float*)d_in[0];
    const float* pos  = (const float*)d_in[1];
    const float* mask = (const float*)d_in[2];
    const float* w1   = (const float*)d_in[3];
    const float* b1   = (const float*)d_in[4];
    const float* w2   = (const float*)d_in[5];
    const float* b2   = (const float*)d_in[6];

    const int B = 8;
    const int N = in_sizes[2] / B;
    int topk = (out_size - B * IC * HW) / (B * 3);
    if (topk <= 0) topk = 500;

    char* ws = (char*)d_ws;
    auto al = [](size_t x) { return (x + 255) & ~(size_t)255; };
    const size_t padBytes = (size_t)B * PH * PW * IC * 2;    // 61,046,784 each
    size_t o_fsum = 0;
    size_t o_cnt  = al(o_fsum + (size_t)B * HW * IC * 4);   // fsum: 117,964,800 B
    size_t o_hist = al(o_cnt  + (size_t)B * HW * 4);        // cnt:  460,800 B
    size_t o_ccnt = al(o_hist + (size_t)B * 1024 * 4);      // hist: 32,768 B
    size_t o_cs   = al(o_ccnt + 256);
    size_t o_ci   = al(o_cs   + (size_t)B * CAP * 4);
    size_t o_th   = al(o_ci   + (size_t)B * CAP * 4);
    size_t o_xh   = al(o_th   + 256);                        // Xh padded f16
    size_t o_y1   = al(o_xh   + padBytes);                   // Y1 padded f16 (adjacent: one zero pass)
    size_t o_w1h  = al(o_y1   + padBytes);
    size_t o_w2h  = al(o_w1h  + (size_t)9 * 256 * 256 * 2);
    size_t o_b2   = al(o_w2h  + (size_t)9 * 80 * 256 * 2);
    // overlays inside the (dead-after-normalize) fsum region:
    size_t o_scores = 0;                 // 29,491,200 B  (heat scores)
    size_t o_z      = 33554432;          // 36,864,000 B  (conv2 out, 80-ch f32) fits in fsum span

    float*     fsum   = (float*)(ws + o_fsum);
    float*     cnt    = (float*)(ws + o_cnt);
    int*       hist   = (int*)  (ws + o_hist);
    int*       ccnt   = (int*)  (ws + o_ccnt);
    float*     cs     = (float*)(ws + o_cs);
    int*       ci     = (int*)  (ws + o_ci);
    float*     th     = (float*)(ws + o_th);
    _Float16*  Xh     = (_Float16*)(ws + o_xh);
    _Float16*  Y1h    = (_Float16*)(ws + o_y1);
    _Float16*  w1h    = (_Float16*)(ws + o_w1h);
    _Float16*  w2h    = (_Float16*)(ws + o_w2h);
    float*     b2pad  = (float*)(ws + o_b2);
    float*     scores = (float*)(ws + o_scores);
    float*     Z      = (float*)(ws + o_z);

    float* out_kp   = (float*)d_out;
    float* out_sc   = out_kp + (size_t)B * topk * 2;
    float* out_grid = out_sc + (size_t)B * topk;

    // 1) zero: fsum+cnt, hist+ccnt, Xh+Y1 (halo padding) — all contiguous spans
    {
        size_t n1 = (o_hist - o_fsum) / 4;
        kd_zero_f32<<<(unsigned)((n1 + 255) / 256), 256, 0, stream>>>(fsum, n1);
        size_t n2 = (o_cs - o_hist) / 4;
        kd_zero_f32<<<(unsigned)((n2 + 255) / 256), 256, 0, stream>>>((float*)hist, n2);
        size_t n3 = (o_w1h - o_xh) / 4;
        kd_zero_f32<<<(unsigned)((n3 + 255) / 256), 256, 0, stream>>>((float*)Xh, n3);
    }
    // 2) weight repack
    kd_prep_w1<<<(9 * 256 * 256 + 255) / 256, 256, 0, stream>>>(w1, w1h);
    kd_prep_w2<<<(9 * 80 * 256 + 255) / 256, 256, 0, stream>>>(w2, b2, w2h, b2pad);
    // 3) event scatter (1 wave per event, 8 waves per block)
    {
        int waves = B * N;
        kd_scatter<<<(waves + 7) / 8, 256, 0, stream>>>(feat, pos, mask, fsum, cnt, B, N);
    }
    // 4) normalize -> feature_grid output (NCHW f32) + padded f16 NHWC copy
    {
        size_t n = (size_t)B * HW * IC;
        kd_normalize<<<(unsigned)((n + 255) / 256), 256, 0, stream>>>(fsum, cnt, out_grid, Xh, B);
    }
    // 5) conv1 (WMMA f16, ReLU) -> padded Y1h; waves = 8*900*16 = 115200 (÷8 exact)
    kd_conv3x3_wmma<256, true, true, true><<<B * 900 * 16 / 8, 256, 0, stream>>>(Xh, w1h, b1, Y1h, B);
    // 6) conv2 (WMMA f16 -> f32, OC padded 80); waves = 8*900*5 = 36000
    kd_conv3x3_wmma<80, false, false, false><<<B * 900 * 5 / 8, 256, 0, stream>>>(Y1h, w2h, b2pad, Z, B);
    // 7) grouped softmax -> heat scores
    kd_heat<<<(B * HW + 255) / 256, 256, 0, stream>>>(Z, scores, B);
    // 8) top-k: histogram -> threshold -> compact -> per-batch bitonic sort
    {
        size_t n = (size_t)B * H8 * W8;
        kd_hist<<<(unsigned)((n + 255) / 256), 256, 0, stream>>>(scores, hist, B);
        kd_thresh<<<1, 32, 0, stream>>>(hist, th, B, topk);
        kd_compact<<<(unsigned)((n + 255) / 256), 256, 0, stream>>>(scores, th, ccnt, cs, ci, B);
        kd_topk<<<B, 512, 0, stream>>>(ccnt, cs, ci, out_kp, out_sc, topk);
    }
}
